// CSPLayer_266287972900
// MI455X (gfx1250) — matrix operand order repaired and checked by reading, not compiled, not run on hardware
//
#include <hip/hip_runtime.h>
#include <math.h>

// ---------------------------------------------------------------------------
// CSPLayer for MI455X (gfx1250): bf16 WMMA (v_wmma_f32_16x16x32_bf16),
// fused per-tile edge/node MLPs, L2-resident gathers, f32 atomic scatter.
// ---------------------------------------------------------------------------

#define Nn 50000
#define Ee 800000
#define Gg 128
#define Hh 128

typedef __attribute__((ext_vector_type(16))) __bf16          v16bf;
typedef __attribute__((ext_vector_type(8)))  float           v8f;
typedef __attribute__((ext_vector_type(8)))  unsigned short  u16x8;
typedef __attribute__((ext_vector_type(16))) unsigned short  u16x16;

static __device__ __forceinline__ v16bf as_bf16(u16x16 u) {
  union { u16x16 u; v16bf b; } c; c.u = u; return c.b;
}

static __device__ __forceinline__ unsigned short f2bf(float x) {
  union { float f; unsigned u; } c; c.f = x;
  unsigned u = c.u;
  u += 0x7FFFu + ((u >> 16) & 1u);   // round to nearest even
  return (unsigned short)(u >> 16);
}

static __device__ __forceinline__ float silu(float x) {
  return x / (1.0f + __expf(-x));
}

// --------------------------- prep kernels ----------------------------------

__global__ void zero_kernel(float* __restrict__ p, int n) {
  int i = blockIdx.x * 256 + threadIdx.x;
  if (i < n) p[i] = 0.0f;
}

// LayerNorm row per wave; writes bf16 h table (L2-resident, 12.8 MB).
__global__ __launch_bounds__(256) void ln_kernel(
    const float* __restrict__ x, const float* __restrict__ gamma,
    const float* __restrict__ beta, unsigned short* __restrict__ h) {
  int wave = (blockIdx.x * blockDim.x + threadIdx.x) >> 5;
  int lane = threadIdx.x & 31;
  if (wave >= Nn) return;
  const float4* row = (const float4*)(x + (size_t)wave * Hh);
  float4 v = row[lane];
  float s  = v.x + v.y + v.z + v.w;
  float s2 = v.x*v.x + v.y*v.y + v.z*v.z + v.w*v.w;
  #pragma unroll
  for (int off = 16; off > 0; off >>= 1) {
    s  += __shfl_xor(s,  off, 32);
    s2 += __shfl_xor(s2, off, 32);
  }
  float mu  = s  * (1.0f / Hh);
  float var = s2 * (1.0f / Hh) - mu * mu;
  float rs  = rsqrtf(var + 1e-5f);
  int c = lane * 4;
  unsigned short o0 = f2bf((v.x - mu) * rs * gamma[c+0] + beta[c+0]);
  unsigned short o1 = f2bf((v.y - mu) * rs * gamma[c+1] + beta[c+1]);
  unsigned short o2 = f2bf((v.z - mu) * rs * gamma[c+2] + beta[c+2]);
  unsigned short o3 = f2bf((v.w - mu) * rs * gamma[c+3] + beta[c+3]);
  uint2 pk; pk.x = (unsigned)o0 | ((unsigned)o1 << 16);
            pk.y = (unsigned)o2 | ((unsigned)o3 << 16);
  *(uint2*)(h + (size_t)wave * Hh + c) = pk;
}

// lat_ip[g,i,k] = row_i(L_g) . row_k(L_g), stored bf16 [G,9]
__global__ void lat_kernel(const float* __restrict__ L,
                           unsigned short* __restrict__ lat9) {
  int idx = blockIdx.x * 256 + threadIdx.x;
  if (idx >= Gg * 9) return;
  int g = idx / 9, r = idx % 9, i = r / 3, k = r % 3;
  const float* a = L + g * 9 + i * 3;
  const float* b = L + g * 9 + k * 3;
  lat9[idx] = f2bf(a[0]*b[0] + a[1]*b[1] + a[2]*b[2]);
}

// Swizzle W[K,128] f32 -> per-lane WMMA B-fragment layout (bf16, K padded).
// Consumption: lane l (n=l&15, g=l>>4), chunk c, n-tile t, elem e reads
// contiguous 16 ushorts at ((c*8+t)*32+l)*16  == W[c*32+(e<8 ? g*8+e : 16+g*8+e-8)][t*16+n].
__global__ void swz_kernel(const float* __restrict__ W,
                           unsigned short* __restrict__ out, int K, int total) {
  int idx = blockIdx.x * 256 + threadIdx.x;
  if (idx >= total) return;
  int e = idx & 15;
  int l = (idx >> 4) & 31;
  int t = (idx >> 9) & 7;
  int c = idx >> 12;
  int n = l & 15, g = l >> 4;
  int k = c * 32 + (e < 8 ? g * 8 + e : 16 + g * 8 + (e - 8));
  int col = t * 16 + n;
  float v = (k < K) ? W[(size_t)k * Hh + col] : 0.0f;
  out[idx] = f2bf(v);
}

__global__ void cnt_kernel(const long long* __restrict__ src,
                           float* __restrict__ cnt) {
  int i = blockIdx.x * 256 + threadIdx.x;
  if (i < Ee) atomicAdd(&cnt[(int)src[i]], 1.0f);
}

// --------------------------- edge kernel -----------------------------------
// 4 waves/block, one 16-edge tile per wave. A tile 16x288 staged in LDS.
// Layer1: 9 K-chunks x 8 N-tiles = 72 WMMAs; Layer2: 4 x 8 = 32 WMMAs.
__global__ __launch_bounds__(128) void edge_kernel(
    const unsigned short* __restrict__ h,
    const float* __restrict__ frac,
    const unsigned short* __restrict__ lat9,
    const long long* __restrict__ srcI,
    const long long* __restrict__ dstI,
    const long long* __restrict__ e2g,
    const unsigned short* __restrict__ W1,
    const float* __restrict__ b1,
    const unsigned short* __restrict__ W2,
    const float* __restrict__ b2,
    float* __restrict__ seg_sum) {
  __shared__ __align__(32) unsigned short Abuf[4][16][288];
  __shared__ int srcbuf[4][16];
  __shared__ int dstbuf[4][16];

  int wave = threadIdx.x >> 5;
  int lane = threadIdx.x & 31;
  int tile = blockIdx.x * 4 + wave;          // grid is exact: E/16 tiles
  int m = lane & 15, g = lane >> 4;

  if (lane < 16) {
    long long e = (long long)tile * 16 + lane;
    int s = (int)srcI[e];
    int d = (int)dstI[e];
    srcbuf[wave][lane] = s;
    dstbuf[wave][lane] = d;
    int gph = (int)e2g[e];
    #pragma unroll
    for (int j = 0; j < 9; ++j) Abuf[wave][lane][256 + j] = lat9[gph * 9 + j];
    #pragma unroll
    for (int j = 0; j < 3; ++j) {
      float f = frac[d * 3 + j] - frac[s * 3 + j];
      f -= floorf(f);                        // (a-b) mod 1.0
      Abuf[wave][lane][265 + j] = f2bf(f);
    }
    #pragma unroll
    for (int j = 268; j < 288; ++j) Abuf[wave][lane][j] = 0;  // K padding
  }
  __syncthreads();

  // Stage hi (cols 0..127) and hj (cols 128..255): 8B per lane per row.
  for (int r = 0; r < 16; ++r) {
    int s = srcbuf[wave][r];
    int d = dstbuf[wave][r];
    const uint2* hs = (const uint2*)(h + (size_t)s * Hh);
    const uint2* hd = (const uint2*)(h + (size_t)d * Hh);
    *(uint2*)&Abuf[wave][r][lane * 4]        = hs[lane];
    *(uint2*)&Abuf[wave][r][128 + lane * 4]  = hd[lane];
  }

  const unsigned short* arow = &Abuf[wave][m][0];

  // ---- Layer 1: [16x288] x [288x128] ----
  v8f acc[8] = {};
  #pragma unroll
  for (int c = 0; c < 9; ++c) {
    u16x8 alo = *(const u16x8*)(arow + c * 32 + g * 8);        // K = c*32+g*8..+7
    u16x8 ahi = *(const u16x8*)(arow + c * 32 + 16 + g * 8);   // K = +16
    u16x16 a = __builtin_shufflevector(alo, ahi, 0,1,2,3,4,5,6,7,8,9,10,11,12,13,14,15);
    #pragma unroll
    for (int t = 0; t < 8; ++t) {
      u16x16 b = *(const u16x16*)(W1 + ((size_t)((c * 8 + t) * 32 + lane) << 4));
      acc[t] = __builtin_amdgcn_wmma_f32_16x16x32_bf16(
          false, as_bf16(a), false, as_bf16(b), (short)0, acc[t], false, false);
    }
  }

  // bias + SiLU, re-stage as bf16 A-tile (cols 0..127) for layer 2
  #pragma unroll
  for (int t = 0; t < 8; ++t) {
    float bias = b1[t * 16 + m];
    #pragma unroll
    for (int r = 0; r < 8; ++r) {
      int row = r + 8 * g;                   // C/D layout rows
      float x = acc[t][r] + bias;
      Abuf[wave][row][t * 16 + m] = f2bf(silu(x));
    }
  }

  // ---- Layer 2: [16x128] x [128x128] ----
  v8f acc2[8] = {};
  #pragma unroll
  for (int c = 0; c < 4; ++c) {
    u16x8 alo = *(const u16x8*)(arow + c * 32 + g * 8);
    u16x8 ahi = *(const u16x8*)(arow + c * 32 + 16 + g * 8);
    u16x16 a = __builtin_shufflevector(alo, ahi, 0,1,2,3,4,5,6,7,8,9,10,11,12,13,14,15);
    #pragma unroll
    for (int t = 0; t < 8; ++t) {
      u16x16 b = *(const u16x16*)(W2 + ((size_t)((c * 8 + t) * 32 + lane) << 4));
      acc2[t] = __builtin_amdgcn_wmma_f32_16x16x32_bf16(
          false, as_bf16(a), false, as_bf16(b), (short)0, acc2[t], false, false);
    }
  }

  // bias + SiLU + scatter-add by src (f32 atomics into L2)
  #pragma unroll
  for (int t = 0; t < 8; ++t) {
    float bias = b2[t * 16 + m];
    int col = t * 16 + m;
    #pragma unroll
    for (int r = 0; r < 8; ++r) {
      int row = r + 8 * g;
      float x = silu(acc2[t][r] + bias);
      int s = srcbuf[wave][row];
      atomicAdd(seg_sum + (size_t)s * Hh + col, x);
    }
  }
}

// --------------------------- node kernel -----------------------------------
// 4 waves/block, one 16-node tile per wave. A = [h | seg/cnt] 16x256.
__global__ __launch_bounds__(128) void node_kernel(
    const unsigned short* __restrict__ h,
    const float* __restrict__ seg_sum,
    const float* __restrict__ cnt,
    const unsigned short* __restrict__ W1,
    const float* __restrict__ b1,
    const unsigned short* __restrict__ W2,
    const float* __restrict__ b2,
    const float* __restrict__ x_in,
    float* __restrict__ out, int nTiles) {
  __shared__ __align__(32) unsigned short Abuf[4][16][256];
  int wave = threadIdx.x >> 5;
  int lane = threadIdx.x & 31;
  int tile = blockIdx.x * 4 + wave;
  if (tile >= nTiles) return;
  int m = lane & 15, g = lane >> 4;

  for (int r = 0; r < 16; ++r) {
    int node = tile * 16 + r;                // N divisible by 16: always valid
    const uint2* hs = (const uint2*)(h + (size_t)node * Hh);
    *(uint2*)&Abuf[wave][r][lane * 4] = hs[lane];
    float inv = 1.0f / fmaxf(cnt[node], 1.0f);
    const float4* ss = (const float4*)(seg_sum + (size_t)node * Hh);
    float4 v = ss[lane];
    unsigned short p0 = f2bf(v.x * inv), p1 = f2bf(v.y * inv);
    unsigned short p2 = f2bf(v.z * inv), p3 = f2bf(v.w * inv);
    uint2 pk; pk.x = (unsigned)p0 | ((unsigned)p1 << 16);
              pk.y = (unsigned)p2 | ((unsigned)p3 << 16);
    *(uint2*)&Abuf[wave][r][128 + lane * 4] = pk;
  }

  const unsigned short* arow = &Abuf[wave][m][0];

  // ---- Layer 1: [16x256] x [256x128] ----
  v8f acc[8] = {};
  #pragma unroll
  for (int c = 0; c < 8; ++c) {
    u16x8 alo = *(const u16x8*)(arow + c * 32 + g * 8);
    u16x8 ahi = *(const u16x8*)(arow + c * 32 + 16 + g * 8);
    u16x16 a = __builtin_shufflevector(alo, ahi, 0,1,2,3,4,5,6,7,8,9,10,11,12,13,14,15);
    #pragma unroll
    for (int t = 0; t < 8; ++t) {
      u16x16 b = *(const u16x16*)(W1 + ((size_t)((c * 8 + t) * 32 + lane) << 4));
      acc[t] = __builtin_amdgcn_wmma_f32_16x16x32_bf16(
          false, as_bf16(a), false, as_bf16(b), (short)0, acc[t], false, false);
    }
  }

  #pragma unroll
  for (int t = 0; t < 8; ++t) {
    float bias = b1[t * 16 + m];
    #pragma unroll
    for (int r = 0; r < 8; ++r) {
      int row = r + 8 * g;
      Abuf[wave][row][t * 16 + m] = f2bf(silu(acc[t][r] + bias));
    }
  }

  // ---- Layer 2: [16x128] x [128x128] ----
  v8f acc2[8] = {};
  #pragma unroll
  for (int c = 0; c < 4; ++c) {
    u16x8 alo = *(const u16x8*)(arow + c * 32 + g * 8);
    u16x8 ahi = *(const u16x8*)(arow + c * 32 + 16 + g * 8);
    u16x16 a = __builtin_shufflevector(alo, ahi, 0,1,2,3,4,5,6,7,8,9,10,11,12,13,14,15);
    #pragma unroll
    for (int t = 0; t < 8; ++t) {
      u16x16 b = *(const u16x16*)(W2 + ((size_t)((c * 8 + t) * 32 + lane) << 4));
      acc2[t] = __builtin_amdgcn_wmma_f32_16x16x32_bf16(
          false, as_bf16(a), false, as_bf16(b), (short)0, acc2[t], false, false);
    }
  }

  // residual: out = node_input + SiLU(acc2 + b2)
  #pragma unroll
  for (int t = 0; t < 8; ++t) {
    float bias = b2[t * 16 + m];
    int col = t * 16 + m;
    #pragma unroll
    for (int r = 0; r < 8; ++r) {
      int row = r + 8 * g;
      int node = tile * 16 + row;
      float x = silu(acc2[t][r] + bias);
      out[(size_t)node * Hh + col] = x_in[(size_t)node * Hh + col] + x;
    }
  }
}

// ------------------------------ launcher -----------------------------------

extern "C" void kernel_launch(void* const* d_in, const int* in_sizes, int n_in,
                              void* d_out, int out_size, void* d_ws, size_t ws_size,
                              hipStream_t stream) {
  const float*     node_features = (const float*)d_in[0];
  const float*     frac    = (const float*)d_in[1];
  const float*     lat     = (const float*)d_in[2];
  const long long* eidx    = (const long long*)d_in[3];
  const long long* e2g     = (const long long*)d_in[4];
  const float*     We1     = (const float*)d_in[5];
  const float*     be1     = (const float*)d_in[6];
  const float*     We2     = (const float*)d_in[7];
  const float*     be2     = (const float*)d_in[8];
  const float*     Wn1     = (const float*)d_in[9];
  const float*     bn1     = (const float*)d_in[10];
  const float*     Wn2     = (const float*)d_in[11];
  const float*     bn2     = (const float*)d_in[12];
  const float*     gamma   = (const float*)d_in[13];
  const float*     beta    = (const float*)d_in[14];
  float*           out     = (float*)d_out;

  // workspace carve-out (256B aligned)
  char* ws = (char*)d_ws;
  size_t cur = 0;
  auto carve = [&](size_t bytes) -> char* {
    char* p = ws + cur;
    cur = (cur + bytes + 255) & ~(size_t)255;
    return p;
  };
  unsigned short* h_bf  = (unsigned short*)carve((size_t)Nn * Hh * 2);   // 12.8 MB
  float*          seg   = (float*)carve((size_t)Nn * Hh * 4);            // 25.6 MB
  float*          cnt   = (float*)carve((size_t)Nn * 4);                 // 0.2 MB
  unsigned short* lat9  = (unsigned short*)carve((size_t)Gg * 9 * 2);
  unsigned short* W1s   = (unsigned short*)carve((size_t)288 * Hh * 2);
  unsigned short* W2s   = (unsigned short*)carve((size_t)128 * Hh * 2);
  unsigned short* Wn1s  = (unsigned short*)carve((size_t)256 * Hh * 2);
  unsigned short* Wn2s  = (unsigned short*)carve((size_t)128 * Hh * 2);

  // 1) zero atomic accumulators (must happen every call)
  {
    int n1 = Nn * Hh;
    zero_kernel<<<(n1 + 255) / 256, 256, 0, stream>>>(seg, n1);
    zero_kernel<<<(Nn + 255) / 256, 256, 0, stream>>>(cnt, Nn);
  }
  // 2) LayerNorm -> bf16 h
  ln_kernel<<<(Nn * 32 + 255) / 256, 256, 0, stream>>>(node_features, gamma, beta, h_bf);
  // 3) L L^T per graph
  lat_kernel<<<(Gg * 9 + 255) / 256, 256, 0, stream>>>(lat, lat9);
  // 4) weight swizzles (bf16, B-fragment layout)
  swz_kernel<<<(288 * Hh + 255) / 256, 256, 0, stream>>>(We1, W1s, 277, 288 * Hh);
  swz_kernel<<<(128 * Hh + 255) / 256, 256, 0, stream>>>(We2, W2s, 128, 128 * Hh);
  swz_kernel<<<(256 * Hh + 255) / 256, 256, 0, stream>>>(Wn1, Wn1s, 256, 256 * Hh);
  swz_kernel<<<(128 * Hh + 255) / 256, 256, 0, stream>>>(Wn2, Wn2s, 128, 128 * Hh);
  // 5) degree counts
  cnt_kernel<<<(Ee + 255) / 256, 256, 0, stream>>>(eidx, cnt);
  // 6) edge MLP + scatter (E/16 = 50000 tiles, 4 waves/block)
  edge_kernel<<<(Ee / 16) / 4, 128, 0, stream>>>(
      h_bf, frac, lat9, eidx, eidx + Ee, e2g, W1s, be1, W2s, be2, seg);
  // 7) node MLP + residual (N/16 = 3125 tiles)
  {
    int nTiles = Nn / 16;
    node_kernel<<<(nTiles + 3) / 4, 128, 0, stream>>>(
        h_bf, seg, cnt, Wn1s, bn1, Wn2s, bn2, node_features, out, nTiles);
  }
  (void)in_sizes; (void)n_in; (void)out_size; (void)ws_size;
}
